// OrthogonalLinear_51573967290709
// MI455X (gfx1250) — matrix operand order repaired and checked by reading, not compiled, not run on hardware
//
#include <hip/hip_runtime.h>

typedef __attribute__((ext_vector_type(16))) __bf16 v16bf;
typedef __attribute__((ext_vector_type(8)))  float  v8f;

union frag_u { unsigned u[8]; v16bf v; };

// f32 pair -> packed bf16 (a -> low 16, b -> high 16).
// Round-half-up (+0x8000) then take high halves with a single v_perm_b32:
// dst = { b.byte3, b.byte2, a.byte3, a.byte2 }  => selector 0x07060302
__device__ __forceinline__ unsigned pk_bf16(float a, float b) {
    unsigned ua = __float_as_uint(a) + 0x8000u;
    unsigned ub = __float_as_uint(b) + 0x8000u;
    return __builtin_amdgcn_perm(ub, ua, 0x07060302u);
}

// exchange lane halves within wave32 (SWAPX16): xor_mask=0x10, and_mask=0x1f
__device__ __forceinline__ float swz16(float x) {
    return __int_as_float(__builtin_amdgcn_ds_swizzle(__float_as_int(x), 0x401F));
}

// ---------------------------------------------------------------------------
// Prep: W = -2 * V^T   (Y = V^T is read directly from V where needed)
// ---------------------------------------------------------------------------
__global__ __launch_bounds__(256) void k_initW(float* __restrict__ W,
                                               const float* __restrict__ V) {
    int t = blockIdx.x * 256 + threadIdx.x;          // 1024*1024 threads
    int r = t >> 10, c = t & 1023;
    W[t] = -2.0f * V[c * 1024 + r];                  // W[r][c] = -2 V[c][r]
}

// ---------------------------------------------------------------------------
// One WY butterfly merge step (reference: m1 = Y_e W_o^T ; W_o += m1^T W_e)
// One workgroup per (even,odd) group pair. Launched 4x (k2 = 1,2,4,8).
// ---------------------------------------------------------------------------
__global__ __launch_bounds__(256) void k_merge(float* __restrict__ W,
                                               const float* __restrict__ V, int k2) {
    __shared__ float M[64];                          // k2*k2 <= 64
    const int p = blockIdx.x, tid = threadIdx.x;
    const int e0 = 2 * p * k2, o0 = (2 * p + 1) * k2;
    for (int idx = tid; idx < k2 * k2; idx += 256) {
        int a = idx / k2, b = idx % k2;
        const float* wrow = W + (size_t)(o0 + b) * 1024;
        int yr = e0 + a;                             // Y[yr][d] = V[d][yr]
        float s = 0.f;
        for (int d = 0; d < 1024; ++d) s += V[(size_t)d * 1024 + yr] * wrow[d];
        M[idx] = s;                                  // M[a*k2+b]
    }
    __syncthreads();
    for (int e = tid; e < k2 * 1024; e += 256) {
        int b = e >> 10, dc = e & 1023;
        float s = 0.f;
        for (int a = 0; a < k2; ++a) s += M[a * k2 + b] * W[(size_t)(e0 + a) * 1024 + dc];
        W[(size_t)(o0 + b) * 1024 + dc] += s;
    }
}

// ---------------------------------------------------------------------------
// Bake Y_i blocks as bf16 WMMA A-fragments (M=16 block rows, K=32 d-chunk).
// Fragment f = i*32+chunk: 32 lanes x 8 dwords, contiguous (coalesced b128).
// ---------------------------------------------------------------------------
__global__ __launch_bounds__(256) void k_yfrag(unsigned* __restrict__ yf,
                                               const float* __restrict__ V) {
    int t = blockIdx.x * 256 + threadIdx.x;          // 64*32*32 threads
    int lane = t & 31, fc = t >> 5;
    int chunk = fc & 31, i = fc >> 5;
    int hi = lane >> 4, row = lane & 15;
    int m = i * 16 + row;                            // Y row = block row
    unsigned* dst = yf + (size_t)fc * 256 + lane * 8;
#pragma unroll
    for (int j = 0; j < 8; ++j) {
        int k = 2 * j + 16 * hi;
        int kk = chunk * 32 + k;                     // Y[m][kk] = V[kk][m]
        dst[j] = pk_bf16(V[(size_t)kk * 1024 + m], V[(size_t)(kk + 1) * 1024 + m]);
    }
}

// ---------------------------------------------------------------------------
// Bake W_i^T d-tiles as bf16 A-fragments (M=16 permuted d rows, K=16 valid,
// K=16..31 zero-padded in-kernel -> only 4 dwords/lane stored).
// M-row <-> d uses the in-register chunk permutation d = cg*32+hi*16+q*8+r.
// ---------------------------------------------------------------------------
__global__ __launch_bounds__(256) void k_wfrag(unsigned* __restrict__ wf,
                                               const float* __restrict__ W) {
    int t = blockIdx.x * 256 + threadIdx.x;          // 64*64*32 threads
    int lane = t & 31, ft = t >> 5;
    int tt = ft & 63, i = ft >> 6;
    int hi = lane >> 4, row = lane & 15;
    int cg = tt >> 1, q = tt & 1;
    int d = cg * 32 + (row >> 3) * 16 + q * 8 + (row & 7);
    unsigned* dst = wf + (size_t)ft * 128 + lane * 4;
#pragma unroll
    for (int j = 0; j < 4; ++j) {
        int k = 2 * j + 8 * hi;                      // W^T[d][k] = W[i*16+k][d]
        dst[j] = pk_bf16(W[(size_t)(i * 16 + k) * 1024 + d],
                         W[(size_t)(i * 16 + k + 1) * 1024 + d]);
    }
}

// ---------------------------------------------------------------------------
// Main: apply 64 sequential rank-16 updates to a 16-batch-column tile.
// 4 waves/WG, each wave owns a 256-row d-slice of X in 16 f32 D-tiles.
// ---------------------------------------------------------------------------
__global__ __launch_bounds__(128) void k_apply(const float* __restrict__ Xin,
                                               float* __restrict__ Xout,
                                               const unsigned* __restrict__ yf,
                                               const unsigned* __restrict__ wf) {
    __shared__ float4 red[2][4][32][2];              // parity-double-buffered T partials
    const int tid = threadIdx.x, lane = tid & 31, w = tid >> 5;
    const int col = lane & 15, hi = lane >> 4;
    const size_t rowbase = (size_t)(blockIdx.x * 16 + col) * 1024;

    v8f x[16];                                       // 128 VGPRs: X master (f32)
#pragma unroll
    for (int tl = 0; tl < 16; ++tl) {
        const int cl = tl >> 1, q = tl & 1;
        const int dbase = w * 256 + cl * 32 + hi * 16 + q * 8;
        const float4* p = (const float4*)(Xin + rowbase + dbase);
        float4 p0 = p[0], p1 = p[1];
        v8f t;
        t[0] = p0.x; t[1] = p0.y; t[2] = p0.z; t[3] = p0.w;
        t[4] = p1.x; t[5] = p1.y; t[6] = p1.z; t[7] = p1.w;
        x[tl] = t;
    }

    for (int i = 63; i >= 0; --i) {                  // blocks applied last-to-first
        const int ip = (i > 0) ? i - 1 : 0;          // prefetch next block's fragments
        __builtin_prefetch(yf + (((size_t)ip * 32 + w * 8) * 32 + lane) * 8, 0, 0);
        __builtin_prefetch(wf + (((size_t)ip * 64 + w * 16) * 32 + lane) * 4, 0, 0);

        // ---- T-stage: partial T(16x16) = Y_i[:, slice] * X[slice, :] ----
        v8f tp;
#pragma unroll
        for (int j = 0; j < 8; ++j) tp[j] = 0.f;
#pragma unroll
        for (int cl = 0; cl < 8; ++cl) {
            const int cg = w * 8 + cl;
            frag_u a, b1;
            const uint4* yp = (const uint4*)(yf + (((size_t)i * 32 + cg) * 32 + lane) * 8);
            uint4 y0 = yp[0], y1 = yp[1];
            a.u[0] = y0.x; a.u[1] = y0.y; a.u[2] = y0.z; a.u[3] = y0.w;
            a.u[4] = y1.x; a.u[5] = y1.y; a.u[6] = y1.z; a.u[7] = y1.w;
            const v8f A = x[2 * cl];                 // chunk rows 0-7 / 16-23
            const v8f B = x[2 * cl + 1];             // chunk rows 8-15 / 24-31
#pragma unroll
            for (int j = 0; j < 4; ++j) b1.u[j]     = pk_bf16(A[2 * j], A[2 * j + 1]);
#pragma unroll
            for (int j = 0; j < 4; ++j) b1.u[4 + j] = pk_bf16(B[2 * j], B[2 * j + 1]);
            tp = __builtin_amdgcn_wmma_f32_16x16x32_bf16(false, a.v, false, b1.v,
                                                         (short)0, tp, false, false);
        }

        // ---- cross-wave reduction of T over the 4 d-slices ----
        const int par = i & 1;
        red[par][w][lane][0] = make_float4(tp[0], tp[1], tp[2], tp[3]);
        red[par][w][lane][1] = make_float4(tp[4], tp[5], tp[6], tp[7]);
        __syncthreads();
        v8f ts;
#pragma unroll
        for (int j = 0; j < 8; ++j) ts[j] = 0.f;
#pragma unroll
        for (int ww = 0; ww < 4; ++ww) {
            float4 r0 = red[par][ww][lane][0];
            float4 r1 = red[par][ww][lane][1];
            ts[0] += r0.x; ts[1] += r0.y; ts[2] += r0.z; ts[3] += r0.w;
            ts[4] += r1.x; ts[5] += r1.y; ts[6] += r1.z; ts[7] += r1.w;
        }

        // ---- T (f32 D-layout) -> bf16 B-fragment, K=16 padded to 32 ----
        frag_u b2;
        const bool lo16 = (lane < 16);
#pragma unroll
        for (int j = 0; j < 4; ++j) {                // K rows 8*hi..8*hi+7: own half
            unsigned v = pk_bf16(ts[2 * j], ts[2 * j + 1]);
            b2.u[j] = v;
        }
#pragma unroll
        for (int j = 0; j < 4; ++j) {                // K rows 16-31: zero pad
            float s0 = swz16(ts[2 * j]);
            float s1 = swz16(ts[2 * j + 1]);
            unsigned v = pk_bf16(s0, s1);
            b2.u[4 + j] = 0u;
        }

        // ---- update: X[slice,:] += W_i^T[slice,:] * T ----
#pragma unroll
        for (int tl = 0; tl < 16; ++tl) {
            frag_u a2;
            const uint4* wp =
                (const uint4*)(wf + (((size_t)i * 64 + w * 16 + tl) * 32 + lane) * 4);
            uint4 w0 = wp[0];
            a2.u[0] = w0.x; a2.u[1] = w0.y; a2.u[2] = w0.z; a2.u[3] = w0.w;
            a2.u[4] = 0u;   a2.u[5] = 0u;   a2.u[6] = 0u;   a2.u[7] = 0u;  // K 16..31 pad
            x[tl] = __builtin_amdgcn_wmma_f32_16x16x32_bf16(false, a2.v, false, b2.v,
                                                            (short)0, x[tl], false, false);
        }
    }

#pragma unroll
    for (int tl = 0; tl < 16; ++tl) {
        const int cl = tl >> 1, q = tl & 1;
        const int dbase = w * 256 + cl * 32 + hi * 16 + q * 8;
        float4* p = (float4*)(Xout + rowbase + dbase);
        v8f t = x[tl];
        p[0] = make_float4(t[0], t[1], t[2], t[3]);
        p[1] = make_float4(t[4], t[5], t[6], t[7]);
    }
}

// ---------------------------------------------------------------------------
extern "C" void kernel_launch(void* const* d_in, const int* in_sizes, int n_in,
                              void* d_out, int out_size, void* d_ws, size_t ws_size,
                              hipStream_t stream) {
    (void)in_sizes; (void)n_in; (void)out_size; (void)ws_size;
    const float* x     = (const float*)d_in[0];   // (8192, 1024)
    const float* V_in  = (const float*)d_in[1];   // (1024, 1024)
    const float* V_out = (const float*)d_in[2];   // (1024, 1024)
    float* out = (float*)d_out;                   // (8192, 1024)

    // workspace layout (dwords): W_in 1M | W_out 1M | yf_in 512K | yf_out 512K
    //                            | wf_in 256K | wf_out 256K   => 14 MB total
    float* W_in  = (float*)d_ws;
    float* W_out = W_in + (1u << 20);
    unsigned* yf_in  = (unsigned*)(W_out + (1u << 20));
    unsigned* yf_out = yf_in + 524288u;
    unsigned* wf_in  = yf_out + 524288u;
    unsigned* wf_out = wf_in + 262144u;

    k_initW<<<4096, 256, 0, stream>>>(W_in, V_in);
    k_initW<<<4096, 256, 0, stream>>>(W_out, V_out);
    for (int k2 = 1; k2 <= 8; k2 <<= 1) {          // 4 sequential merge steps
        int npairs = 1024 / (2 * k2);
        k_merge<<<npairs, 256, 0, stream>>>(W_in, V_in, k2);
        k_merge<<<npairs, 256, 0, stream>>>(W_out, V_out, k2);
    }
    k_yfrag<<<256, 256, 0, stream>>>(yf_in, V_in);
    k_yfrag<<<256, 256, 0, stream>>>(yf_out, V_out);
    k_wfrag<<<512, 256, 0, stream>>>(wf_in, W_in);
    k_wfrag<<<512, 256, 0, stream>>>(wf_out, W_out);

    k_apply<<<512, 128, 0, stream>>>(x, out, yf_in, wf_in);     // input layer
    k_apply<<<512, 128, 0, stream>>>(out, out, yf_out, wf_out); // output layer
}
